// VQVAE_7275674600260
// MI455X (gfx1250) — matrix operand order repaired and checked by reading, not compiled, not run on hardware
//
#include <hip/hip_runtime.h>
#include <hip/hip_bf16.h>
#include <stdint.h>

// ---------------- Types ----------------
typedef __bf16 bf16_t;
typedef __attribute__((ext_vector_type(16))) __bf16 v16bf;
typedef __attribute__((ext_vector_type(8)))  float  v8f;

union frag16 { uint4 u[2]; v16bf v; };

#define BATCH     8192
#define NFRAMES   25
#define FRAME     10
#define FEAT      512
#define EMBED     128
#define NCODES    512
#define NPOS      (BATCH * NFRAMES)       // 204800
#define LRELU_A   0.2f
#define LN_EPS    1e-5f

// ---------------- Device helpers ----------------
__device__ inline float lrelu(float x) { return x > 0.f ? x : LRELU_A * x; }

// Block reduction (256 threads, 8 waves). op: 0=sum, 1=min
__device__ inline float blk_reduce(float v, float* red8, int op) {
  int lane = threadIdx.x & 31, wv = threadIdx.x >> 5;
#pragma unroll
  for (int o = 16; o > 0; o >>= 1) {
    float t = __shfl_xor(v, o);
    v = op ? fminf(v, t) : (v + t);
  }
  if (lane == 0) red8[wv] = v;
  __syncthreads();
  if (wv == 0) {
    float t = (lane < 8) ? red8[lane] : (op ? 3.4e38f : 0.f);
#pragma unroll
    for (int o = 4; o > 0; o >>= 1) {
      float u = __shfl_xor(t, o);
      t = op ? fminf(t, u) : (t + u);
    }
    if (lane == 0) red8[0] = t;
  }
  __syncthreads();
  float r = red8[0];
  __syncthreads();
  return r;
}

// ---------------- Weight swizzle: f32 (K,N) row-major -> bf16 [K/32][N][32] -----------
__global__ void k_wswz(const float* __restrict__ in, bf16_t* __restrict__ out,
                       int K, int N) {
  int total = K * N;
  for (int i = blockIdx.x * blockDim.x + threadIdx.x; i < total;
       i += gridDim.x * blockDim.x) {
    int kk = i & 31;
    int r  = i >> 5;          // kc*N + n
    int n  = r % N;
    int kc = r / N;
    out[i] = (bf16_t)in[(size_t)(kc * 32 + kk) * N + n];
  }
}

// ---------------- Frame encoder: (B,250) -> (B*25,512) with LN+lrelu ----------------
__global__ __launch_bounds__(256) void k_frame_encode(
    const float* __restrict__ x, const float* __restrict__ w, const float* __restrict__ bias,
    const float* __restrict__ g, const float* __restrict__ be,
    float* __restrict__ yf, bf16_t* __restrict__ yb) {
  __shared__ float xr[16];
  __shared__ float red8[8];
  int p = blockIdx.x;
  int bb = p / NFRAMES, t = p % NFRAMES;
  int tid = threadIdx.x;
  if (tid < FRAME) xr[tid] = x[(size_t)bb * (NFRAMES * FRAME) + t * FRAME + tid];
  __syncthreads();
  float v0 = bias[tid], v1 = bias[tid + 256];
#pragma unroll
  for (int k = 0; k < FRAME; ++k) {
    float xv = xr[k];
    v0 += xv * w[k * FEAT + tid];
    v1 += xv * w[k * FEAT + tid + 256];
  }
  float s  = blk_reduce(v0 + v1, red8, 0);
  float s2 = blk_reduce(v0 * v0 + v1 * v1, red8, 0);
  float mean = s * (1.f / FEAT);
  float var  = s2 * (1.f / FEAT) - mean * mean;
  float rstd = rsqrtf(var + LN_EPS);
  float o0 = lrelu((v0 - mean) * rstd * g[tid] + be[tid]);
  float o1 = lrelu((v1 - mean) * rstd * g[tid + 256] + be[tid + 256]);
  size_t base = (size_t)p * FEAT;
  yf[base + tid] = o0;         yf[base + tid + 256] = o1;
  yb[base + tid] = (bf16_t)o0; yb[base + tid + 256] = (bf16_t)o1;
}

// ---------------- TDNN fragment helpers ----------------
__device__ inline void load_bfrags(frag16* bfr, const bf16_t* __restrict__ wswz,
                                   int kc48, int colg, int m, int kbB) {
  const bf16_t* wp = wswz + ((size_t)kc48 * FEAT + colg * 64 + m) * 32 + kbB;
#pragma unroll
  for (int ct = 0; ct < 4; ++ct) {
    bfr[ct].u[0] = *(const uint4*)(wp + (size_t)ct * 16 * 32);
    bfr[ct].u[1] = *(const uint4*)(wp + (size_t)ct * 16 * 32 + 16);
  }
}

__device__ inline void load_afrags(frag16* af, const bf16_t* __restrict__ xb,
                                   const int* bbv, const int* ttv, int kc48, int kbA) {
  const int dt = kc48 >> 4, kc = kc48 & 15;
  const uint4 z = {0u, 0u, 0u, 0u};
#pragma unroll
  for (int rt = 0; rt < 4; ++rt) {
    int ts = ttv[rt] + dt - 1;
    af[rt].u[0] = z; af[rt].u[1] = z;
    if (ts >= 0 && ts < NFRAMES) {
      const bf16_t* ap = xb + ((size_t)(bbv[rt] * NFRAMES + ts)) * FEAT + kc * 32 + kbA;
      af[rt].u[0] = *(const uint4*)ap;
      af[rt].u[1] = *(const uint4*)(ap + 16);
    }
  }
}

__device__ inline void wmma_all(v8f acc[4][4], frag16* af, frag16* bfr) {
#pragma unroll
  for (int ct = 0; ct < 4; ++ct)
#pragma unroll
    for (int rt = 0; rt < 4; ++rt)
      acc[rt][ct] = __builtin_amdgcn_wmma_f32_16x16x32_bf16(
          false, af[rt].v, false, bfr[ct].v, (short)0, acc[rt][ct], false, false);
}

// ---------------- TDNN block: y = x + lrelu(LN(conv_k3(x) + b)) ----------------
// 512 threads = 16 waves; block tile = 128 rows x 512 cols.
// Software-pipelined K-loop: two B-fragment buffers; B(k+1) loads are in flight
// while the 16 WMMAs for B(k) execute.
__global__ __launch_bounds__(512) void k_tdnn(
    const float* __restrict__ xf, const bf16_t* __restrict__ xb,
    const bf16_t* __restrict__ wswz,
    const float* __restrict__ bias, const float* __restrict__ g, const float* __restrict__ be,
    float* __restrict__ yf, bf16_t* __restrict__ yb) {
  __shared__ float conv[32 * FEAT];    // 64 KB, reused over 4 epilogue passes

  const int tid  = threadIdx.x;
  const int wave = tid >> 5, lane = tid & 31;
  const int rowg = wave & 1;      // rows rowg*64 .. +63
  const int colg = wave >> 1;     // cols colg*64 .. +63
  const int pos0 = blockIdx.x * 128;

  const int m    = lane & 15;
  const int half = lane >> 4;
  const int kbA  = half ? 8 : 0;
  const int kbB  = half ? 8 : 0;

  int bbv[4], ttv[4];
#pragma unroll
  for (int rt = 0; rt < 4; ++rt) {
    int prow = pos0 + rowg * 64 + rt * 16 + m;
    bbv[rt] = prow / NFRAMES;
    ttv[rt] = prow % NFRAMES;
  }

  v8f acc[4][4];
#pragma unroll
  for (int rt = 0; rt < 4; ++rt)
#pragma unroll
    for (int ct = 0; ct < 4; ++ct)
#pragma unroll
      for (int v = 0; v < 8; ++v) acc[rt][ct][v] = 0.f;

  frag16 b0[4], b1[4], af[4];
  load_bfrags(b0, wswz, 0, colg, m, kbB);
  for (int kc = 0; kc < 48; kc += 2) {
    load_afrags(af, xb, bbv, ttv, kc, kbA);
    load_bfrags(b1, wswz, kc + 1, colg, m, kbB);
    wmma_all(acc, af, b0);                       // b1 loads in flight
    load_afrags(af, xb, bbv, ttv, kc + 1, kbA);
    load_bfrags(b0, wswz, (kc + 2 < 48) ? kc + 2 : 47, colg, m, kbB);
    wmma_all(acc, af, b1);                       // b0 loads in flight
  }

  // Epilogue: 4 passes of 32 rows through the 64KB conv buffer
  for (int pass = 0; pass < 4; ++pass) {
    __syncthreads();
    if (rowg == (pass >> 1)) {
      int rtBase = (pass & 1) * 2;
#pragma unroll
      for (int rr = 0; rr < 2; ++rr) {
        int rt = rtBase + rr;
#pragma unroll
        for (int ct = 0; ct < 4; ++ct) {
          int n = colg * 64 + ct * 16 + m;
          float bv = bias[n];
#pragma unroll
          for (int v = 0; v < 8; ++v)
            conv[(rr * 16 + half * 8 + v) * FEAT + n] = acc[rt][ct][v] + bv;
        }
      }
    }
    __syncthreads();
    for (int r = wave; r < 32; r += 16) {
      float s = 0.f, s2 = 0.f;
      for (int i = lane; i < FEAT; i += 32) {
        float v = conv[r * FEAT + i];
        s += v; s2 += v * v;
      }
#pragma unroll
      for (int o = 16; o > 0; o >>= 1) { s += __shfl_xor(s, o); s2 += __shfl_xor(s2, o); }
      float mean = s * (1.f / FEAT);
      float var  = s2 * (1.f / FEAT) - mean * mean;
      float rstd = rsqrtf(var + LN_EPS);
      size_t base = (size_t)(pos0 + pass * 32 + r) * FEAT;
      for (int i = lane; i < FEAT; i += 32) {
        float c = conv[r * FEAT + i];
        float l = lrelu((c - mean) * rstd * g[i] + be[i]);
        float o = xf[base + i] + l;
        yf[base + i] = o;
        yb[base + i] = (bf16_t)o;
      }
    }
  }
}

// ---------------- Generic dense GEMM (bf16 WMMA, direct-global fragments) ---------
template <int K, int N, bool DO_LN, bool DO_LRELU, bool WRITE_BF>
__global__ __launch_bounds__(256) void k_dense(
    const bf16_t* __restrict__ xb, const bf16_t* __restrict__ wswz,  // [K/32][N][32]
    const float* __restrict__ bias, const float* __restrict__ g, const float* __restrict__ be,
    float* __restrict__ yf, bf16_t* __restrict__ yb) {
  constexpr int TPW = N / 64;      // tiles per wave (2 or 8)
  __shared__ float conv[32 * N];

  const int tid  = threadIdx.x;
  const int wave = tid >> 5, lane = tid & 31;
  const int wrow = wave & 1;
  const int wcol = wave >> 1;
  const int pos0 = blockIdx.x * 32;

  const int m    = lane & 15;
  const int half = lane >> 4;
  const int kbA  = half ? 8 : 0;
  const int kbB  = half ? 8 : 0;
  const int prow = pos0 + wrow * 16 + m;

  v8f acc[TPW];
#pragma unroll
  for (int t8 = 0; t8 < TPW; ++t8)
#pragma unroll
    for (int v = 0; v < 8; ++v) acc[t8][v] = 0.f;

#pragma unroll
  for (int kc = 0; kc < K / 32; ++kc) {
    frag16 a;
    const bf16_t* ap = xb + (size_t)prow * K + kc * 32 + kbA;
    a.u[0] = *(const uint4*)ap;
    a.u[1] = *(const uint4*)(ap + 16);
    frag16 bfr[TPW];
    const bf16_t* wp = wswz + ((size_t)kc * N + wcol * (N / 4) + m) * 32 + kbB;
#pragma unroll
    for (int t8 = 0; t8 < TPW; ++t8) {
      bfr[t8].u[0] = *(const uint4*)(wp + (size_t)t8 * 16 * 32);
      bfr[t8].u[1] = *(const uint4*)(wp + (size_t)t8 * 16 * 32 + 16);
    }
#pragma unroll
    for (int t8 = 0; t8 < TPW; ++t8)
      acc[t8] = __builtin_amdgcn_wmma_f32_16x16x32_bf16(
          false, a.v, false, bfr[t8].v, (short)0, acc[t8], false, false);
  }
#pragma unroll
  for (int t8 = 0; t8 < TPW; ++t8) {
    int n = wcol * (N / 4) + t8 * 16 + m;
    float bv = bias[n];
#pragma unroll
    for (int v = 0; v < 8; ++v) {
      int mm = wrow * 16 + half * 8 + v;
      conv[mm * N + n] = acc[t8][v] + bv;
    }
  }
  __syncthreads();
  for (int r = wave; r < 32; r += 8) {
    float mean = 0.f, rstd = 1.f;
    if (DO_LN) {
      float s = 0.f, s2 = 0.f;
      for (int i = lane; i < N; i += 32) {
        float v = conv[r * N + i];
        s += v; s2 += v * v;
      }
#pragma unroll
      for (int o = 16; o > 0; o >>= 1) { s += __shfl_xor(s, o); s2 += __shfl_xor(s2, o); }
      mean = s * (1.f / N);
      float var = s2 * (1.f / N) - mean * mean;
      rstd = rsqrtf(var + LN_EPS);
    }
    size_t base = (size_t)(pos0 + r) * N;
    for (int i = lane; i < N; i += 32) {
      float c = conv[r * N + i];
      float o = DO_LN ? ((c - mean) * rstd * g[i] + be[i]) : c;
      if (DO_LRELU) o = lrelu(o);
      yf[base + i] = o;
      if (WRITE_BF) yb[base + i] = (bf16_t)o;
    }
  }
}

// ---------------- Pool: LN per (b,t) row then mean over t -> (B,512) -------------
__global__ __launch_bounds__(256) void k_pool_ln(
    const float* __restrict__ h, const float* __restrict__ g, const float* __restrict__ be,
    float* __restrict__ yf, bf16_t* __restrict__ yb) {
  __shared__ float red8[8];
  int bb = blockIdx.x, tid = threadIdx.x;
  float a0 = 0.f, a1 = 0.f;
  for (int t = 0; t < NFRAMES; ++t) {
    size_t base = ((size_t)bb * NFRAMES + t) * FEAT;
    float v0 = h[base + tid], v1 = h[base + tid + 256];
    float s  = blk_reduce(v0 + v1, red8, 0);
    float s2 = blk_reduce(v0 * v0 + v1 * v1, red8, 0);
    float mean = s * (1.f / FEAT);
    float var  = s2 * (1.f / FEAT) - mean * mean;
    float rstd = rsqrtf(var + LN_EPS);
    a0 += (v0 - mean) * rstd * g[tid] + be[tid];
    a1 += (v1 - mean) * rstd * g[tid + 256] + be[tid + 256];
  }
  a0 *= (1.f / NFRAMES); a1 *= (1.f / NFRAMES);
  size_t ob = (size_t)bb * FEAT;
  yf[ob + tid] = a0;         yf[ob + tid + 256] = a1;
  yb[ob + tid] = (bf16_t)a0; yb[ob + tid + 256] = (bf16_t)a1;
}

// ---------------- VQ: distances, softmax cls, masked argmin, z_q, partial losses -----
__global__ __launch_bounds__(256) void k_vq(
    const float* __restrict__ ze, const float* __restrict__ cb, const int* __restrict__ labels,
    float* __restrict__ zq_out, float* __restrict__ idx_out, bf16_t* __restrict__ zq_bf,
    float* __restrict__ cls_part, float* __restrict__ mse_part) {
  __shared__ float z[EMBED];
  __shared__ float red8[8];
  __shared__ float av[256];
  __shared__ int   ai[256];
  int bb = blockIdx.x, tid = threadIdx.x;
  if (tid < EMBED) z[tid] = ze[(size_t)bb * EMBED + tid];
  __syncthreads();
  float dv[2];
#pragma unroll
  for (int j = 0; j < 2; ++j) {
    int c = tid + j * 256;
    const float* cr = cb + (size_t)c * EMBED;
    float s = 0.f;
    for (int k = 0; k < EMBED; ++k) { float df = z[k] - cr[k]; s += df * df; }
    dv[j] = sqrtf(fmaxf(s, 0.f));
  }
  float mn = blk_reduce(fminf(dv[0], dv[1]), red8, 1);
  float e0 = expf(-(dv[0] - mn)), e1 = expf(-(dv[1] - mn));
  float tot  = blk_reduce(e0 + e1, red8, 0);
  float nsum = blk_reduce(e0, red8, 0);   // codes [0,256) = "normal" set
  int lab = labels[bb];
  if (tid == 0) {
    float p = nsum / tot;
    p = fminf(fmaxf(p, 1e-12f), 1.f - 1e-12f);
    cls_part[bb] = (lab == 0) ? -logf(p) : -log1pf(-p);
  }
  av[tid] = (lab == 0) ? dv[0] : dv[1];
  ai[tid] = (lab == 0) ? tid : tid + 256;
  __syncthreads();
  for (int s = 128; s > 0; s >>= 1) {
    if (tid < s) {
      float v2 = av[tid + s]; int i2 = ai[tid + s];
      if (v2 < av[tid] || (v2 == av[tid] && i2 < ai[tid])) { av[tid] = v2; ai[tid] = i2; }
    }
    __syncthreads();
  }
  int best = ai[0];
  float sq = 0.f;
  if (tid < EMBED) {
    float q = cb[(size_t)best * EMBED + tid];
    zq_out[(size_t)bb * EMBED + tid] = q;
    zq_bf[(size_t)bb * EMBED + tid]  = (bf16_t)q;
    float df = z[tid] - q;
    sq = df * df;
  }
  float tsq = blk_reduce(sq, red8, 0);
  if (tid == 0) { mse_part[bb] = tsq; idx_out[bb] = (float)best; }
}

__global__ __launch_bounds__(256) void k_reduce_loss(
    const float* __restrict__ cls_part, const float* __restrict__ mse_part,
    float* __restrict__ out_cb, float* __restrict__ out_cm, float* __restrict__ out_cls) {
  __shared__ float red8[8];
  float sc = 0.f, sm = 0.f;
  for (int i = threadIdx.x; i < BATCH; i += 256) { sc += cls_part[i]; sm += mse_part[i]; }
  sc = blk_reduce(sc, red8, 0);
  sm = blk_reduce(sm, red8, 0);
  if (threadIdx.x == 0) {
    *out_cls = sc / (float)BATCH;
    float m = sm / ((float)BATCH * (float)EMBED);
    *out_cb = m; *out_cm = m;
  }
}

// ---------------- Broadcast (B,512) -> (B,25,512) + pos, f32 + bf16 -------------
__global__ void k_broadcast_pos(const float* __restrict__ h, const float* __restrict__ pos,
                                float* __restrict__ yf, bf16_t* __restrict__ yb) {
  const size_t total = (size_t)NPOS * FEAT;
  for (size_t i = (size_t)blockIdx.x * blockDim.x + threadIdx.x; i < total;
       i += (size_t)gridDim.x * blockDim.x) {
    size_t c = i % FEAT;
    size_t p = i / FEAT;
    size_t bb = p / NFRAMES, t = p % NFRAMES;
    float o = h[bb * FEAT + c] + pos[t * FEAT + c];
    yf[i] = o;
    yb[i] = (bf16_t)o;
  }
}

// ---------------- Final decoder head: LN then 512 -> 10 ----------------
__global__ __launch_bounds__(256) void k_final_dec(
    const float* __restrict__ h, const float* __restrict__ g, const float* __restrict__ be,
    const float* __restrict__ fdw, const float* __restrict__ fdb, float* __restrict__ out) {
  __shared__ float nb[FEAT];
  __shared__ float red8[8];
  int p = blockIdx.x, tid = threadIdx.x;
  size_t base = (size_t)p * FEAT;
  float v0 = h[base + tid], v1 = h[base + tid + 256];
  float s  = blk_reduce(v0 + v1, red8, 0);
  float s2 = blk_reduce(v0 * v0 + v1 * v1, red8, 0);
  float mean = s * (1.f / FEAT);
  float var  = s2 * (1.f / FEAT) - mean * mean;
  float rstd = rsqrtf(var + LN_EPS);
  nb[tid]       = (v0 - mean) * rstd * g[tid] + be[tid];
  nb[tid + 256] = (v1 - mean) * rstd * g[tid + 256] + be[tid + 256];
  __syncthreads();
  if (tid < 160) {
    int o = tid >> 4, sg = tid & 15;
    float acc = 0.f;
    for (int i = sg; i < FEAT; i += 16) acc += nb[i] * fdw[i * FRAME + o];
#pragma unroll
    for (int off = 8; off > 0; off >>= 1) acc += __shfl_xor(acc, off, 16);
    if (sg == 0) out[(size_t)p * FRAME + o] = acc + fdb[o];
  }
}

// ---------------- Host launch ----------------
extern "C" void kernel_launch(void* const* d_in, const int* in_sizes, int n_in,
                              void* d_out, int out_size, void* d_ws, size_t ws_size,
                              hipStream_t stream) {
  (void)in_sizes; (void)n_in; (void)out_size; (void)ws_size;
  const float* x      = (const float*)d_in[0];
  const int*   labels = (const int*)d_in[1];
  const float* fe_w = (const float*)d_in[2];
  const float* fe_b = (const float*)d_in[3];
  const float* en_g = (const float*)d_in[4];
  const float* en_b = (const float*)d_in[5];
  const float* encW[4], *encB[4], *encG[4], *encBE[4];
  for (int i = 0; i < 4; ++i) {
    encW[i]  = (const float*)d_in[6 + 4 * i + 0];
    encB[i]  = (const float*)d_in[6 + 4 * i + 1];
    encG[i]  = (const float*)d_in[6 + 4 * i + 2];
    encBE[i] = (const float*)d_in[6 + 4 * i + 3];
  }
  const float* fn_g  = (const float*)d_in[22];
  const float* fn_b  = (const float*)d_in[23];
  const float* pre_w = (const float*)d_in[24];
  const float* pre_b = (const float*)d_in[25];
  const float* pn_g  = (const float*)d_in[26];
  const float* pn_b  = (const float*)d_in[27];
  const float* out_w = (const float*)d_in[28];
  const float* out_b = (const float*)d_in[29];
  const float* codebook = (const float*)d_in[30];
  const float* in_w  = (const float*)d_in[31];
  const float* in_b  = (const float*)d_in[32];
  const float* in_g  = (const float*)d_in[33];
  const float* in_be = (const float*)d_in[34];
  const float* ex_w  = (const float*)d_in[35];
  const float* ex_b  = (const float*)d_in[36];
  const float* ex_g  = (const float*)d_in[37];
  const float* ex_be = (const float*)d_in[38];
  const float* pos   = (const float*)d_in[39];
  const float* decW[4], *decB[4], *decG[4], *decBE[4];
  for (int i = 0; i < 4; ++i) {
    decW[i]  = (const float*)d_in[40 + 4 * i + 0];
    decB[i]  = (const float*)d_in[40 + 4 * i + 1];
    decG[i]  = (const float*)d_in[40 + 4 * i + 2];
    decBE[i] = (const float*)d_in[40 + 4 * i + 3];
  }
  const float* dfn_g = (const float*)d_in[56];
  const float* dfn_b = (const float*)d_in[57];
  const float* fd_w  = (const float*)d_in[58];
  const float* fd_b  = (const float*)d_in[59];

  float* out = (float*)d_out;
  const size_t O_XR  = 0;
  const size_t O_ZE  = O_XR + (size_t)BATCH * NFRAMES * FRAME;
  const size_t O_ZQ  = O_ZE + (size_t)BATCH * EMBED;
  const size_t O_IDX = O_ZQ + (size_t)BATCH * EMBED;
  const size_t O_CB  = O_IDX + BATCH;
  const size_t O_CM  = O_CB + 1;
  const size_t O_CLS = O_CM + 1;

  char* wsp = (char*)d_ws;
  auto alloc = [&](size_t bytes) {
    char* p = wsp;
    wsp += (bytes + 255) & ~(size_t)255;
    return p;
  };
  const size_t ACT_F = (size_t)NPOS * FEAT * sizeof(float);
  const size_t ACT_B = (size_t)NPOS * FEAT * sizeof(bf16_t);
  bf16_t* wEnc[4]; for (int i = 0; i < 4; ++i) wEnc[i] = (bf16_t*)alloc(3 * FEAT * FEAT * 2);
  bf16_t* wDec[4]; for (int i = 0; i < 4; ++i) wDec[i] = (bf16_t*)alloc(3 * FEAT * FEAT * 2);
  bf16_t* wPre = (bf16_t*)alloc(FEAT * FEAT * 2);
  bf16_t* wOut = (bf16_t*)alloc(FEAT * EMBED * 2);
  bf16_t* wIn  = (bf16_t*)alloc(EMBED * FEAT * 2);
  bf16_t* wEx  = (bf16_t*)alloc(FEAT * FEAT * 2);
  float*  hAf = (float*)alloc(ACT_F);  bf16_t* hAb = (bf16_t*)alloc(ACT_B);
  float*  hBf = (float*)alloc(ACT_F);  bf16_t* hBb = (bf16_t*)alloc(ACT_B);
  float*  poolF = (float*)alloc((size_t)BATCH * FEAT * 4);
  bf16_t* poolB = (bf16_t*)alloc((size_t)BATCH * FEAT * 2);
  float*  preF  = (float*)alloc((size_t)BATCH * FEAT * 4);
  bf16_t* preB  = (bf16_t*)alloc((size_t)BATCH * FEAT * 2);
  bf16_t* zqB   = (bf16_t*)alloc((size_t)BATCH * EMBED * 2);
  float*  dec1F = (float*)alloc((size_t)BATCH * FEAT * 4);
  bf16_t* dec1B = (bf16_t*)alloc((size_t)BATCH * FEAT * 2);
  float*  dec2F = (float*)alloc((size_t)BATCH * FEAT * 4);
  float*  clsP  = (float*)alloc(BATCH * 4);
  float*  mseP  = (float*)alloc(BATCH * 4);

  auto swz = [&](const float* src, bf16_t* dst, int K, int N) {
    int n = K * N;
    int blk = (n + 1023) / 1024;
    k_wswz<<<blk, 256, 0, stream>>>(src, dst, K, N);
  };
  for (int i = 0; i < 4; ++i) swz(encW[i], wEnc[i], 3 * FEAT, FEAT);
  for (int i = 0; i < 4; ++i) swz(decW[i], wDec[i], 3 * FEAT, FEAT);
  swz(pre_w, wPre, FEAT, FEAT);
  swz(out_w, wOut, FEAT, EMBED);
  swz(in_w,  wIn,  EMBED, FEAT);
  swz(ex_w,  wEx,  FEAT, FEAT);

  // 2) Encoder
  k_frame_encode<<<NPOS, 256, 0, stream>>>(x, fe_w, fe_b, en_g, en_b, hAf, hAb);
  float* cf = hAf; bf16_t* cbh = hAb; float* nf = hBf; bf16_t* nbh = hBb;
  for (int i = 0; i < 4; ++i) {
    k_tdnn<<<NPOS / 128, 512, 0, stream>>>(cf, cbh, wEnc[i], encB[i], encG[i], encBE[i], nf, nbh);
    float* tf = cf; cf = nf; nf = tf;
    bf16_t* tb = cbh; cbh = nbh; nbh = tb;
  }
  k_pool_ln<<<BATCH, 256, 0, stream>>>(cf, fn_g, fn_b, poolF, poolB);
  k_dense<FEAT, FEAT, true, true, true><<<BATCH / 32, 256, 0, stream>>>(
      poolB, wPre, pre_b, pn_g, pn_b, preF, preB);
  k_dense<FEAT, EMBED, false, false, false><<<BATCH / 32, 256, 0, stream>>>(
      preB, wOut, out_b, nullptr, nullptr, out + O_ZE, nullptr);

  // 3) VQ + losses
  k_vq<<<BATCH, 256, 0, stream>>>(out + O_ZE, codebook, labels,
                                  out + O_ZQ, out + O_IDX, zqB, clsP, mseP);
  k_reduce_loss<<<1, 256, 0, stream>>>(clsP, mseP, out + O_CB, out + O_CM, out + O_CLS);

  // 4) Decoder
  k_dense<EMBED, FEAT, true, true, true><<<BATCH / 32, 256, 0, stream>>>(
      zqB, wIn, in_b, in_g, in_be, dec1F, dec1B);
  k_dense<FEAT, FEAT, true, true, false><<<BATCH / 32, 256, 0, stream>>>(
      dec1B, wEx, ex_b, ex_g, ex_be, dec2F, nullptr);
  k_broadcast_pos<<<8192, 256, 0, stream>>>(dec2F, pos, hAf, hAb);
  cf = hAf; cbh = hAb; nf = hBf; nbh = hBb;
  for (int i = 0; i < 4; ++i) {
    k_tdnn<<<NPOS / 128, 512, 0, stream>>>(cf, cbh, wDec[i], decB[i], decG[i], decBE[i], nf, nbh);
    float* tf = cf; cf = nf; nf = tf;
    bf16_t* tb = cbh; cbh = nbh; nbh = tb;
  }
  k_final_dec<<<NPOS, 256, 0, stream>>>(cf, dfn_g, dfn_b, fd_w, fd_b, out + O_XR);
}
